// MultiHeadedAttentionWithDepth_36369783063352
// MI455X (gfx1250) — matrix operand order repaired and checked by reading, not compiled, not run on hardware
//
#include <hip/hip_runtime.h>

typedef __bf16 bf16;
typedef __attribute__((ext_vector_type(16))) __bf16 v16bf;
typedef __attribute__((ext_vector_type(8)))  __bf16 v8bf;
typedef __attribute__((ext_vector_type(8)))  float  v8f;

#define NSEQ 2048
#define NB   4
#define C1D  512
#define C2D  128
#define NHD  8
#define DH   80     // c1h + c2h = 64 + 16
#define DHP  96     // padded to 3 * 32 for K-steps
#define MROWS 8192  // B * N
#define CCAT 640    // C1 + C2

#define KLDS_STRIDE 104  // 96 + 8 pad (bank-conflict-free B-fragment reads)
#define VLDS_STRIDE 72   // 64 + 8 pad
#define KTILE_ELEMS (64 * KLDS_STRIDE)
#define VTILE_ELEMS (DH * VLDS_STRIDE)

__device__ __forceinline__ v8f wmma_bf16(v16bf a, v16bf b, v8f c) {
  return __builtin_amdgcn_wmma_f32_16x16x32_bf16(false, a, false, b, (short)0, c,
                                                 false, false);
}

// A-fragment (16x32 bf16): lane holds row (lane%16); element[0..15] = K kb..kb+15,
// kb = (lane/16)*16.  Two 16B contiguous loads.
__device__ __forceinline__ v16bf ld_afrag(const bf16* __restrict__ p) {
  v8bf lo = *(const v8bf*)(p);
  v8bf hi = *(const v8bf*)(p + 8);
  v16bf a;
#pragma unroll
  for (int i = 0; i < 8; ++i) { a[i] = lo[i]; a[i + 8] = hi[i]; }
  return a;
}

// gfx1250 async global->LDS copy, 16B per lane (ASYNCcnt-tracked path)
__device__ __forceinline__ void async_copy16(unsigned lds_off, const void* gptr) {
  asm volatile("global_load_async_to_lds_b128 %0, %1, off"
               :
               : "v"(lds_off), "v"(gptr)
               : "memory");
}

__device__ __forceinline__ void wait_asynccnt0() {
  asm volatile("s_wait_asynccnt 0x0" ::: "memory");
}

// ---- DPP-based 16-lane-row reductions (pure VALU, no LDS round trips) ----
// EXEC is all-1s in the regions where these are used, so bound_ctrl=1 is safe.

template <int CTRL>
__device__ __forceinline__ float dpp_xchg(float v) {
  return __int_as_float(
      __builtin_amdgcn_update_dpp(0, __float_as_int(v), CTRL, 0xF, 0xF, true));
}

__device__ __forceinline__ float rowmax16(float v) {
  v = fmaxf(v, dpp_xchg<0xB1>(v));   // quad_perm [1,0,3,2]  (xor 1)
  v = fmaxf(v, dpp_xchg<0x4E>(v));   // quad_perm [2,3,0,1]  (xor 2)
  v = fmaxf(v, dpp_xchg<0x141>(v));  // row_half_mirror      (merge quads)
  v = fmaxf(v, dpp_xchg<0x140>(v));  // row_mirror           (merge halves)
  return v;
}

__device__ __forceinline__ float rowsum16(float v) {
  v += dpp_xchg<0xB1>(v);
  v += dpp_xchg<0x4E>(v);
  v += dpp_xchg<0x141>(v);
  v += dpp_xchg<0x140>(v);
  return v;
}

// Cooperative async staging of one 64-key K/V chunk into LDS (per-block).
__device__ __forceinline__ void stage_tiles(int tid, unsigned kbase,
                                            unsigned vbase,
                                            const bf16* __restrict__ Kb,
                                            const bf16* __restrict__ Vb,
                                            int kc) {
  // K chunk: 64 rows x 192B, 12 x 16B units/row (768 units)
#pragma unroll
  for (int it = 0; it < 3; ++it) {
    int u = tid + it * 256;
    int row = u / 12, col = u - row * 12;
    async_copy16(kbase + (unsigned)(row * (KLDS_STRIDE * 2) + col * 16),
                 (const char*)Kb + (size_t)(kc + row) * (DHP * 2) + col * 16);
  }
  // V chunk: 80 rows x 128B, 8 x 16B units/row (640 units)
#pragma unroll
  for (int it = 0; it < 3; ++it) {
    int u = tid + it * 256;
    if (u < 640) {
      int row = u >> 3, col = u & 7;
      async_copy16(vbase + (unsigned)(row * (VLDS_STRIDE * 2) + col * 16),
                   (const char*)Vb + (size_t)row * (NSEQ * 2) +
                       (size_t)kc * 2 + col * 16);
    }
  }
}

// ---------------- elementwise prep kernels ----------------

__global__ void concat_cvt_kernel(const float* __restrict__ color,
                                  const float* __restrict__ depth,
                                  bf16* __restrict__ X, int total) {
  int i = blockIdx.x * blockDim.x + threadIdx.x;
  if (i >= total) return;
  int g = i / CCAT, c = i % CCAT;
  float v = (c < C1D) ? color[(size_t)g * C1D + c]
                      : depth[(size_t)g * C2D + (c - C1D)];
  X[i] = (bf16)v;
}

__global__ void transpose_cvt_kernel(const float* __restrict__ W,
                                     bf16* __restrict__ Wt, int K, int N,
                                     int total) {
  int i = blockIdx.x * blockDim.x + threadIdx.x;
  if (i >= total) return;
  int n = i / K, k = i % K;          // Wt[n][k] = W[k][n]
  Wt[i] = (bf16)W[(size_t)k * N + n];
}

__global__ void zero_bf16_kernel(bf16* __restrict__ p, int total) {
  int i = blockIdx.x * blockDim.x + threadIdx.x;
  if (i < total) p[i] = (bf16)0.0f;
}

// ---------------- WMMA GEMM with mode-specific epilogues ----------------
// Y[row][c] = (sum_k X[row][k] * Wt[c][k] + bias[c]) * oscale
// MODE 0: scatter bf16 into [B,H,N,DHP]   (Q / K projection, dh-padded)
// MODE 1: scatter bf16 into Vt [B,H,DH,N] at d = c%64      (color values)
// MODE 2: scatter bf16 into Vt [B,H,DH,N] at d = 64 + c%16 (depth values)
// MODE 3: plain f32 [row][c] (final output projections)

template <int MODE>
__global__ __launch_bounds__(256) void gemm_kernel(
    const bf16* __restrict__ X, int ldx, const bf16* __restrict__ Wt,
    const float* __restrict__ bias, int K, int Nout, float oscale,
    void* __restrict__ outp) {
  const int lane = threadIdx.x & 31;
  const int w    = threadIdx.x >> 5;
  const int half = lane >> 4;
  const int l16  = lane & 15;
  const int m0 = blockIdx.y * 128 + (w & 3) * 32;   // 4x2 wave grid
  const int n0 = blockIdx.x * 128 + (w >> 2) * 64;  // wave tile: 32x64

  v8f acc[2][4];
#pragma unroll
  for (int i = 0; i < 2; ++i)
#pragma unroll
    for (int j = 0; j < 4; ++j) acc[i][j] = {};

  for (int k = 0; k < K; k += 32) {
    v16bf a[2];
#pragma unroll
    for (int i = 0; i < 2; ++i)
      a[i] = ld_afrag(X + (size_t)(m0 + i * 16 + l16) * ldx + k + half * 16);
#pragma unroll
    for (int j = 0; j < 4; ++j) {
      // B-fragment: lane = out col, 16 consecutive k at (lane/16)*16
      v16bf bb = *(const v16bf*)(Wt + (size_t)(n0 + j * 16 + l16) * K + k + half * 16);
#pragma unroll
      for (int i = 0; i < 2; ++i) acc[i][j] = wmma_bf16(a[i], bb, acc[i][j]);
    }
  }

#pragma unroll
  for (int j = 0; j < 4; ++j) {
    const int c = n0 + j * 16 + l16;
    const float bv = bias[c];
#pragma unroll
    for (int i = 0; i < 2; ++i) {
#pragma unroll
      for (int r = 0; r < 8; ++r) {
        const int row = m0 + i * 16 + r + 8 * half;  // C/D layout row
        const float val = (acc[i][j][r] + bv) * oscale;
        if constexpr (MODE == 0) {
          bf16* out = (bf16*)outp;
          int b = row >> 11, n = row & (NSEQ - 1);
          int hh = c / DH, d = c % DH;
          out[(((size_t)b * NHD + hh) * NSEQ + n) * DHP + d] = (bf16)val;
        } else if constexpr (MODE == 1) {
          bf16* out = (bf16*)outp;
          int b = row >> 11, n = row & (NSEQ - 1);
          int hh = c >> 6, d = c & 63;
          out[(((size_t)b * NHD + hh) * DH + d) * NSEQ + n] = (bf16)val;
        } else if constexpr (MODE == 2) {
          bf16* out = (bf16*)outp;
          int b = row >> 11, n = row & (NSEQ - 1);
          int hh = c >> 4, d = 64 + (c & 15);
          out[(((size_t)b * NHD + hh) * DH + d) * NSEQ + n] = (bf16)val;
        } else {
          ((float*)outp)[(size_t)row * Nout + c] = val;
        }
      }
    }
  }
}

// ---------------- fused flash attention ----------------
// grid.y = b*H+h (32), grid.x = query block of 128 (16). 8 waves/block,
// each wave owns a 16-row Q strip. K/V tiles for each 64-key chunk are staged
// into double-buffered LDS via async global->LDS copies (ASYNCcnt): chunk i+1
// is prefetched while chunk i is computed, hiding the global->LDS latency.
// Q fragments live in registers for the whole kernel; Q is pre-scaled by
// 1/sqrt(80) in its projection epilogue.

__global__ __launch_bounds__(256) void attn_kernel(
    const bf16* __restrict__ Q, const bf16* __restrict__ Kmat,
    const bf16* __restrict__ Vt, bf16* __restrict__ Oc, bf16* __restrict__ Od) {
  __shared__ __align__(16) bf16 kTile[2][KTILE_ELEMS];  // keys x dh(pad)
  __shared__ __align__(16) bf16 vTile[2][VTILE_ELEMS];  // d x keys(pad)
  __shared__ __align__(16) bf16 pTile[8][16][72];       // per-wave P
  const int tid  = threadIdx.x;
  const int lane = tid & 31;
  const int w    = tid >> 5;
  const int half = lane >> 4;
  const int l16  = lane & 15;
  const int bh = blockIdx.y;
  const int b  = bh >> 3;
  const int h  = bh & 7;
  const int q0 = blockIdx.x * 128 + w * 16;

  const bf16* Qb = Q    + (size_t)bh * NSEQ * DHP;
  const bf16* Kb = Kmat + (size_t)bh * NSEQ * DHP;
  const bf16* Vb = Vt   + (size_t)bh * DH * NSEQ;

  const unsigned kLds = (unsigned)(size_t)(void*)&kTile[0][0];
  const unsigned vLds = (unsigned)(size_t)(void*)&vTile[0][0];

  // Q A-fragments for this wave's 16 rows, all 3 K-steps (kept in registers)
  v16bf qa[3];
#pragma unroll
  for (int ks = 0; ks < 3; ++ks)
    qa[ks] = ld_afrag(Qb + (size_t)(q0 + l16) * DHP + ks * 32 + half * 16);

  v8f o[5];
#pragma unroll
  for (int vt = 0; vt < 5; ++vt) o[vt] = {};
  float mrow[8], lrow[8];
#pragma unroll
  for (int r = 0; r < 8; ++r) { mrow[r] = -3.0e38f; lrow[r] = 0.0f; }

  const float LOG2E = 1.4426950408889634f;

  // prologue: stage chunk 0 into buffer 0
  stage_tiles(tid, kLds, vLds, Kb, Vb, 0);

  for (int kc = 0; kc < NSEQ; kc += 64) {
    const int cur = (kc >> 6) & 1;
    wait_asynccnt0();  // current chunk's tiles landed (this wave's share)
    __syncthreads();   // all waves' shares visible; prev reads of other buffer done

    // prefetch next chunk into the other buffer while we compute this one
    if (kc + 64 < NSEQ) {
      stage_tiles(tid, kLds + (unsigned)((cur ^ 1) * (KTILE_ELEMS * 2)),
                  vLds + (unsigned)((cur ^ 1) * (VTILE_ELEMS * 2)), Kb, Vb,
                  kc + 64);
    }

    const bf16* kt = &kTile[cur][0];
    const bf16* vtb = &vTile[cur][0];

    // ---- S = Q K^T for 16x64 tile (4 n-subtiles x 3 k-steps = 12 WMMA) ----
    v8f s[4];
#pragma unroll
    for (int t = 0; t < 4; ++t) s[t] = {};
#pragma unroll
    for (int t = 0; t < 4; ++t) {
      const bf16* kp = kt + (t * 16 + l16) * KLDS_STRIDE + half * 16;
#pragma unroll
      for (int ks = 0; ks < 3; ++ks) {
        v16bf kfrag = *(const v16bf*)(kp + ks * 32);
        s[t] = wmma_bf16(qa[ks], kfrag, s[t]);
      }
    }

    // ---- online softmax: rows live on 16-lane half groups (DPP reduce) ----
    float sf[8];
#pragma unroll
    for (int r = 0; r < 8; ++r) {
      float v = fmaxf(fmaxf(s[0][r], s[1][r]), fmaxf(s[2][r], s[3][r]));
      v = rowmax16(v);
      float mnew = fmaxf(mrow[r], v);
      sf[r] = exp2f((mrow[r] - mnew) * LOG2E);
      mrow[r] = mnew;
    }
#pragma unroll
    for (int t = 0; t < 4; ++t)
#pragma unroll
      for (int r = 0; r < 8; ++r)
        s[t][r] = exp2f((s[t][r] - mrow[r]) * LOG2E);
#pragma unroll
    for (int r = 0; r < 8; ++r) {
      float v = rowsum16(s[0][r] + s[1][r] + s[2][r] + s[3][r]);
      lrow[r] = lrow[r] * sf[r] + v;
    }
#pragma unroll
    for (int vt = 0; vt < 5; ++vt)
#pragma unroll
      for (int r = 0; r < 8; ++r) o[vt][r] *= sf[r];

    // ---- transpose P (C-layout -> A-layout) through per-wave LDS ----
#pragma unroll
    for (int t = 0; t < 4; ++t)
#pragma unroll
      for (int r = 0; r < 8; ++r)
        pTile[w][r + 8 * half][t * 16 + l16] = (bf16)s[t][r];
    __syncthreads();

    // ---- O += P V  (2 k-steps x 5 d-subtiles = 10 WMMA) ----
#pragma unroll
    for (int kst = 0; kst < 2; ++kst) {
      v16bf pa = ld_afrag(&pTile[w][l16][kst * 32 + half * 16]);
#pragma unroll
      for (int vt = 0; vt < 5; ++vt) {
        const bf16* vp =
            vtb + (vt * 16 + l16) * VLDS_STRIDE + kst * 32 + half * 16;
        v16bf vfrag = *(const v16bf*)vp;
        o[vt] = wmma_bf16(pa, vfrag, o[vt]);
      }
    }
  }

#pragma unroll
  for (int r = 0; r < 8; ++r) lrow[r] = 1.0f / lrow[r];

  // scatter O into [B,N,H*64] (color) / [B,N,H*16] (depth) bf16 buffers
#pragma unroll
  for (int vt = 0; vt < 5; ++vt) {
    const int d = vt * 16 + l16;
#pragma unroll
    for (int r = 0; r < 8; ++r) {
      const int row = q0 + r + 8 * half;
      const float val = o[vt][r] * lrow[r];
      const size_t g = (size_t)b * NSEQ + row;
      if (d < 64) Oc[g * C1D + h * 64 + d] = (bf16)val;
      else        Od[g * C2D + h * 16 + (d - 64)] = (bf16)val;
    }
  }
}

// ---------------- host launch ----------------

extern "C" void kernel_launch(void* const* d_in, const int* in_sizes, int n_in,
                              void* d_out, int out_size, void* d_ws,
                              size_t ws_size, hipStream_t stream) {
  (void)in_sizes; (void)n_in; (void)out_size; (void)ws_size;
  const float* color = (const float*)d_in[0];
  const float* depth = (const float*)d_in[1];
  const float* Wq  = (const float*)d_in[2];  const float* bq  = (const float*)d_in[3];
  const float* Wk  = (const float*)d_in[4];  const float* bk  = (const float*)d_in[5];
  const float* Wcv = (const float*)d_in[6];  const float* bcv = (const float*)d_in[7];
  const float* Wdv = (const float*)d_in[8];  const float* bdv = (const float*)d_in[9];
  const float* Wcl = (const float*)d_in[10]; const float* bcl = (const float*)d_in[11];
  const float* Wdl = (const float*)d_in[12]; const float* bdl = (const float*)d_in[13];

  char* ws = (char*)d_ws;
  size_t off = 0;
  auto take = [&](size_t bytes) {
    char* p = ws + off;
    off += (bytes + 255) & ~(size_t)255;
    return p;
  };

  bf16* Xcat = (bf16*)take((size_t)MROWS * CCAT * 2);
  bf16* WqT  = (bf16*)take((size_t)CCAT * CCAT * 2);
  bf16* WkT  = (bf16*)take((size_t)CCAT * CCAT * 2);
  bf16* WcvT = (bf16*)take((size_t)C1D * C1D * 2);
  bf16* WdvT = (bf16*)take((size_t)C2D * C2D * 2);
  bf16* WclT = (bf16*)take((size_t)C1D * C1D * 2);
  bf16* WdlT = (bf16*)take((size_t)C2D * C2D * 2);
  bf16* Qbuf = (bf16*)take((size_t)NB * NHD * NSEQ * DHP * 2);
  bf16* Kbuf = (bf16*)take((size_t)NB * NHD * NSEQ * DHP * 2);
  bf16* Vtb  = (bf16*)take((size_t)NB * NHD * DH * NSEQ * 2);
  bf16* Oc   = (bf16*)take((size_t)MROWS * C1D * 2);
  bf16* Od   = (bf16*)take((size_t)MROWS * C2D * 2);

  const int TB = 256;
  {
    int total = MROWS * CCAT;
    concat_cvt_kernel<<<(total + TB - 1) / TB, TB, 0, stream>>>(color, depth,
                                                                Xcat, total);
  }
  {
    int t1 = CCAT * CCAT;
    transpose_cvt_kernel<<<(t1 + TB - 1) / TB, TB, 0, stream>>>(Wq, WqT, CCAT, CCAT, t1);
    transpose_cvt_kernel<<<(t1 + TB - 1) / TB, TB, 0, stream>>>(Wk, WkT, CCAT, CCAT, t1);
    int t2 = C1D * C1D;
    transpose_cvt_kernel<<<(t2 + TB - 1) / TB, TB, 0, stream>>>(Wcv, WcvT, C1D, C1D, t2);
    transpose_cvt_kernel<<<(t2 + TB - 1) / TB, TB, 0, stream>>>(Wcl, WclT, C1D, C1D, t2);
    int t3 = C2D * C2D;
    transpose_cvt_kernel<<<(t3 + TB - 1) / TB, TB, 0, stream>>>(Wdv, WdvT, C2D, C2D, t3);
    transpose_cvt_kernel<<<(t3 + TB - 1) / TB, TB, 0, stream>>>(Wdl, WdlT, C2D, C2D, t3);
  }
  {
    int total = NB * NHD * NSEQ * DHP;  // zero dh-pad region of Q/K
    zero_bf16_kernel<<<(total + TB - 1) / TB, TB, 0, stream>>>(Qbuf, total);
    zero_bf16_kernel<<<(total + TB - 1) / TB, TB, 0, stream>>>(Kbuf, total);
  }
  const float qscale = 0.11180339887498949f;  // 1/sqrt(80), folded into Q
  // projections (bf16 WMMA GEMMs with layout-scattering epilogues)
  gemm_kernel<0><<<dim3(CCAT / 128, MROWS / 128), TB, 0, stream>>>(
      Xcat, CCAT, WqT, bq, CCAT, CCAT, qscale, Qbuf);
  gemm_kernel<0><<<dim3(CCAT / 128, MROWS / 128), TB, 0, stream>>>(
      Xcat, CCAT, WkT, bk, CCAT, CCAT, 1.0f, Kbuf);
  gemm_kernel<1><<<dim3(C1D / 128, MROWS / 128), TB, 0, stream>>>(
      Xcat, CCAT, WcvT, bcv, C1D, C1D, 1.0f, Vtb);
  gemm_kernel<2><<<dim3(C2D / 128, MROWS / 128), TB, 0, stream>>>(
      Xcat + C1D, CCAT, WdvT, bdv, C2D, C2D, 1.0f, Vtb);
  // fused flash attention
  attn_kernel<<<dim3(NSEQ / 128, NB * NHD), TB, 0, stream>>>(Qbuf, Kbuf, Vtb,
                                                             Oc, Od);
  // final linears straight into d_out (f32)
  float* out_color = (float*)d_out;
  float* out_depth = out_color + (size_t)MROWS * C1D;
  gemm_kernel<3><<<dim3(C1D / 128, MROWS / 128), TB, 0, stream>>>(
      Oc, C1D, WclT, bcl, C1D, C1D, 1.0f, out_color);
  gemm_kernel<3><<<dim3(C2D / 128, MROWS / 128), TB, 0, stream>>>(
      Od, C2D, WdlT, bdl, C2D, C2D, 1.0f, out_depth);
}
